// SpixEmbedding_25082609009343
// MI455X (gfx1250) — matrix operand order repaired and checked by reading, not compiled, not run on hardware
//
#include <hip/hip_runtime.h>
#include <stdint.h>

typedef __attribute__((ext_vector_type(16))) _Float16 v16h;
typedef __attribute__((ext_vector_type(8)))  float    v8f;
typedef __attribute__((ext_vector_type(4)))  unsigned int v4u;

#define B_    8
#define C_    16
#define H_    256
#define W_    256
#define E_    384
#define P_    256
#define PSZ_  16
#define NSEG_ 256
#define KTOT  (PSZ_*PSZ_*C_)   // 4096
#define NKC   (KTOT/32)        // 128 K-chunks of 32
#define NNT   (E_/16)          // 24 N-tiles of 16

union AB32 {
    v16h v;
    _Float16 h[16];
    v4u q[2];
};

struct W4 { _Float16 h[4]; };   // 8 bytes: packed corner weights (flag folded in)

// ---------------- bbox kernels ----------------
__global__ void k_init_bbox(int* __restrict__ bb) {
    int i = blockIdx.x * blockDim.x + threadIdx.x;
    int n = B_ * NSEG_;
    if (i < n) {
        bb[0*n + i] = 0x7fffffff;  // rmin
        bb[1*n + i] = -1;          // rmax
        bb[2*n + i] = 0x7fffffff;  // cmin
        bb[3*n + i] = -1;          // cmax
    }
}

__global__ void k_bbox(const int* __restrict__ seg, int* __restrict__ bb) {
    int idx = blockIdx.x * blockDim.x + threadIdx.x;
    if (idx >= B_ * H_ * W_) return;
    int c = idx % W_;
    int r = (idx / W_) % H_;
    int b = idx / (H_ * W_);
    int s = seg[idx];
    int n = B_ * NSEG_;
    int o = b * NSEG_ + s;
    atomicMin(&bb[0*n + o], r);
    atomicMax(&bb[1*n + o], r);
    atomicMin(&bb[2*n + o], c);
    atomicMax(&bb[3*n + o], c);
}

// ---------------- W -> f16 B-fragment pre-swizzle ----------------
// Wh[((kc*NNT + nt)*32 + lane)*8 + vreg] = packed {f16(K), f16(K+1)}
// K = kc*32 + (lane>>4)*16 + 2*vreg, N = nt*16 + (lane&15)
// (contiguous k order, same as the A fragment: lanes 0-15 K=0-15, lanes 16-31 K=16-31)
__global__ void k_packW(const float* __restrict__ Wm, uint32_t* __restrict__ Wh) {
    int tid = blockIdx.x * blockDim.x + threadIdx.x;
    if (tid >= NKC * NNT * 32 * 8) return;
    int vreg = tid & 7;
    int lane = (tid >> 3) & 31;
    int g    = tid >> 8;            // kc*NNT + nt
    int nt   = g % NNT;
    int kc   = g / NNT;
    int K = kc * 32 + (lane >> 4) * 16 + 2 * vreg;
    int N = nt * 16 + (lane & 15);
    float lo = Wm[(size_t)K * E_ + N];
    float hi = Wm[(size_t)(K + 1) * E_ + N];
    union { uint32_t u; _Float16 h[2]; } p;
    p.h[0] = (_Float16)lo;
    p.h[1] = (_Float16)hi;
    Wh[tid] = p.u;
}

// ---------------- fused sample + WMMA GEMM ----------------
__global__ void __launch_bounds__(256) k_fused(
    const float* __restrict__ x, const int* __restrict__ seg,
    const int* __restrict__ crop, const uint32_t* __restrict__ Wh,
    const int* __restrict__ bb, float* __restrict__ out)
{
    __shared__ float fra[256], fca[256];
    __shared__ int   r0a[256], c0a[256];
    __shared__ int   sid[16];
    __shared__ W4    wpre[4096];           // 32 KB: all (pk,pl,m) corner weights
    __shared__ float featLDS[2][16 * 34];  // double-buffered A panel, stride 34

    const int t    = threadIdx.x;
    const int blk  = blockIdx.x;
    const int b    = blk >> 4;             // batch
    const int pt   = blk & 15;             // patch tile (16 patches)
    const int lane = t & 31;
    const int wave = t >> 5;
    const int nbb  = B_ * NSEG_;
    const int nt0  = wave * 3;             // 8 waves x 3 N-tiles = 24

    // ---- coordinate precompute: one thread per (patch m, k) ----
    {
        int m = t >> 4, k = t & 15;
        int id = crop[b * P_ + pt * 16 + m];
        if (k == 0) sid[m] = id;
        int rmn = bb[0*nbb + b*NSEG_ + id];
        int rmx = bb[1*nbb + b*NSEG_ + id];
        int cmn = bb[2*nbb + b*NSEG_ + id];
        int cmx = bb[3*nbb + b*NSEG_ + id];
        int extr = rmx - rmn + 1, extc = cmx - cmn + 1;
        bool issmall = (extr <= PSZ_) && (extc <= PSZ_);
        int r0i, c0i; float frf, fcf;
        if (issmall) {
            r0i = rmn + k; frf = 0.f;
            c0i = cmn + k; fcf = 0.f;
        } else {
            float er = (float)extr;
            float offr = fminf(fmaxf(((float)k + 0.5f) * er / 16.f - 0.5f, 0.f), er - 1.f);
            float rf = (float)rmn + offr;
            r0i = (int)floorf(rf); frf = rf - (float)r0i;
            float ec = (float)extc;
            float offc = fminf(fmaxf(((float)k + 0.5f) * ec / 16.f - 0.5f, 0.f), ec - 1.f);
            float cf = (float)cmn + offc;
            c0i = (int)floorf(cf); fcf = cf - (float)c0i;
        }
        r0a[t] = r0i; fra[t] = frf;
        c0a[t] = c0i; fca[t] = fcf;
    }
    __syncthreads();

    // ---- weight/mask precompute for all 4096 (pk,pl,m) triples ----
    // Clamped unconditional seg loads + selects (no exec-mask branches).
    {
        const int* sb = seg + (size_t)b * H_ * W_;
        #pragma unroll
        for (int it = 0; it < 16; ++it) {
            int tri = it * 256 + t;
            int m  = tri & 15;
            int pl = (tri >> 4) & 15;
            int pk = tri >> 8;
            int r0 = r0a[m*16 + pk]; float fr = fra[m*16 + pk];
            int c0 = c0a[m*16 + pl]; float fc = fca[m*16 + pl];
            int id = sid[m];
            int rc0 = min(r0,     H_ - 1), rc1 = min(r0 + 1, H_ - 1);
            int cl0 = min(c0,     W_ - 1), cl1 = min(c0 + 1, W_ - 1);
            bool vr0 = r0     < H_, vr1 = r0 + 1 < H_;
            bool vc0 = c0     < W_, vc1 = c0 + 1 < W_;
            int s00 = sb[rc0*W_ + cl0];
            int s01 = sb[rc0*W_ + cl1];
            int s10 = sb[rc1*W_ + cl0];
            int s11 = sb[rc1*W_ + cl1];
            float m00 = (vr0 && vc0 && s00 == id) ? 1.f : 0.f;
            float m01 = (vr0 && vc1 && s01 == id) ? 1.f : 0.f;
            float m10 = (vr1 && vc0 && s10 == id) ? 1.f : 0.f;
            float m11 = (vr1 && vc1 && s11 == id) ? 1.f : 0.f;
            float w00 = (1.f - fr) * (1.f - fc), w01 = (1.f - fr) * fc;
            float w10 = fr * (1.f - fc),         w11 = fr * fc;
            float segf = w00*m00 + w01*m01 + w10*m10 + w11*m11;
            float flag = (segf != 0.f) ? 1.f : 0.f;
            W4 w;
            w.h[0] = (_Float16)(w00 * flag);
            w.h[1] = (_Float16)(w01 * flag);
            w.h[2] = (_Float16)(w10 * flag);
            w.h[3] = (_Float16)(w11 * flag);
            wpre[tri] = w;
        }
    }

    const float* xb0 = x + (size_t)b * C_ * H_ * W_;

    // produce: bilinear-gather chunk kc into buf (each thread: 1 triple, 2 channels)
    auto produce = [&](int kc, float* buf) {
        int pk  = kc >> 3;
        int pl0 = (kc & 7) * 2;
        int m   = t >> 4;
        int sub = t & 15;
        int j   = sub >> 3;
        int cch = (sub & 7) * 2;
        int pl  = pl0 + j;
        int tri = (pk << 8) + (pl << 4) + m;
        W4 w = wpre[tri];
        float w00 = (float)w.h[0], w01 = (float)w.h[1];
        float w10 = (float)w.h[2], w11 = (float)w.h[3];
        int r0 = r0a[m*16 + pk];
        int c0 = c0a[m*16 + pl];
        int rc0 = min(r0,     H_ - 1), rc1 = min(r0 + 1, H_ - 1);
        int cl0 = min(c0,     W_ - 1), cl1 = min(c0 + 1, W_ - 1);
        bool vr0 = r0     < H_, vr1 = r0 + 1 < H_;
        bool vc0 = c0     < W_, vc1 = c0 + 1 < W_;
        bool v00 = vr0 && vc0, v01 = vr0 && vc1;
        bool v10 = vr1 && vc0, v11 = vr1 && vc1;
        int i00 = rc0*W_ + cl0, i01 = rc0*W_ + cl1;
        int i10 = rc1*W_ + cl0, i11 = rc1*W_ + cl1;
        float vals[2];
        #pragma unroll
        for (int cc = 0; cc < 2; ++cc) {
            const float* xb = xb0 + (size_t)(cch + cc) * H_ * W_;
            float x00 = xb[i00]; x00 = v00 ? x00 : 0.f;
            float x01 = xb[i01]; x01 = v01 ? x01 : 0.f;
            float x10 = xb[i10]; x10 = v10 ? x10 : 0.f;
            float x11 = xb[i11]; x11 = v11 ? x11 : 0.f;
            vals[cc] = w00*x00 + w01*x01 + w10*x10 + w11*x11;
        }
        *(float2*)&buf[m*34 + j*16 + cch] = make_float2(vals[0], vals[1]);
    };

    // consume: A fragment from buf (16-bit A 16x32 layout) + 3x WMMA
    v8f acc[3];
    acc[0] = (v8f){}; acc[1] = (v8f){}; acc[2] = (v8f){};
    auto consume = [&](int kc, const float* buf) {
        AB32 afrag;
        int mm = lane & 15, half = lane >> 4;
        #pragma unroll
        for (int vr = 0; vr < 8; ++vr) {
            int k0 = half * 16 + vr * 2;
            afrag.h[2*vr]   = (_Float16)buf[mm*34 + k0];
            afrag.h[2*vr+1] = (_Float16)buf[mm*34 + k0 + 1];
        }
        #pragma unroll
        for (int q = 0; q < 3; ++q) {
            int nt = nt0 + q;
            AB32 bfrag;
            const v4u* wp = (const v4u*)(Wh + (((size_t)(kc * NNT + nt)) * 32 + lane) * 8);
            bfrag.q[0] = wp[0];
            bfrag.q[1] = wp[1];
            acc[q] = __builtin_amdgcn_wmma_f32_16x16x32_f16(
                false, afrag.v, false, bfrag.v, (short)0, acc[q], false, false);
        }
    };

    __syncthreads();                 // wpre ready
    produce(0, featLDS[0]);
    __syncthreads();                 // chunk 0 panel ready

    #pragma unroll 2
    for (int kc = 0; kc < NKC; ++kc) {
        int cur = kc & 1;
        if (kc + 1 < NKC) produce(kc + 1, featLDS[cur ^ 1]);  // gathers overlap WMMA
        consume(kc, featLDS[cur]);
        __syncthreads();             // next panel written; current panel reads done
    }

    // ---- store C tiles (16x16 f32 C/D layout) ----
    {
        int half = lane >> 4, n = lane & 15;
        int prow = b * P_ + pt * 16;
        #pragma unroll
        for (int q = 0; q < 3; ++q) {
            int N = (nt0 + q) * 16 + n;
            #pragma unroll
            for (int i = 0; i < 8; ++i) {
                int M = half * 8 + i;
                out[(size_t)(prow + M) * E_ + N] = acc[q][i];
            }
        }
    }
}

extern "C" void kernel_launch(void* const* d_in, const int* in_sizes, int n_in,
                              void* d_out, int out_size, void* d_ws, size_t ws_size,
                              hipStream_t stream) {
    const float* x    = (const float*)d_in[0];
    const int*   seg  = (const int*)d_in[1];
    const int*   crop = (const int*)d_in[2];
    const float* Wm   = (const float*)d_in[3];
    float*       out  = (float*)d_out;

    int*      bb = (int*)d_ws;                                   // 4*B*NSEG ints = 32 KB
    uint32_t* Wh = (uint32_t*)((char*)d_ws + (size_t)4 * B_ * NSEG_ * sizeof(int)); // 3 MB

    k_init_bbox<<<(B_ * NSEG_ + 255) / 256, 256, 0, stream>>>(bb);
    k_bbox<<<(B_ * H_ * W_ + 255) / 256, 256, 0, stream>>>(seg, bb);
    k_packW<<<(NKC * NNT * 32 * 8 + 255) / 256, 256, 0, stream>>>(Wm, Wh);
    k_fused<<<B_ * (P_ / 16), 256, 0, stream>>>(x, seg, crop, Wh, bb, out);
}
